// GCN_LSTM_84507776516237
// MI455X (gfx1250) — compile-verified, matrix operand order checked by reading, not run on hardware
//
#include <hip/hip_runtime.h>
#include <hip/hip_bf16.h>
#include <math.h>

// ---------------------------------------------------------------------------
// CDNA5 (gfx1250) GCN + BiLSTM pipeline, round 2.
// Dense GEMMs: v_wmma_f32_16x16x32_f16, 2 N-tiles per wave, K fully unrolled,
// zero per-element predication (padded activations + pre-packed f16 weights).
// ---------------------------------------------------------------------------

typedef __attribute__((ext_vector_type(16))) _Float16 v16h;
typedef __attribute__((ext_vector_type(8)))  float    v8f;

// ---------------- degree / norm ----------------

__global__ void init_deg_kernel(float* deg_out, float* deg_in, int N) {
  int i = blockIdx.x * blockDim.x + threadIdx.x;
  if (i < N) { deg_out[i] = 1.0f; deg_in[i] = 1.0f; }  // self-loop counted
}

__global__ void edge_deg_kernel(const int* __restrict__ src,
                                const int* __restrict__ dst,
                                float* deg_out, float* deg_in, int E) {
  int e = blockIdx.x * blockDim.x + threadIdx.x;
  if (e < E) {
    atomicAdd(&deg_out[src[e]], 1.0f);
    atomicAdd(&deg_in[dst[e]], 1.0f);
  }
}

__global__ void finalize_nrm_kernel(float* d, int N) {
  int i = blockIdx.x * blockDim.x + threadIdx.x;
  if (i < N) d[i] = rsqrtf(fmaxf(d[i], 1.0f));
}

// ---------------- h0 = [x | emb[node_id] | zero-pad to 64] ----------------

__global__ void build_h0_kernel(const float* __restrict__ x,
                                const int* __restrict__ node_id,
                                const float* __restrict__ emb,
                                float* __restrict__ h0, int N) {
  int idx = blockIdx.x * blockDim.x + threadIdx.x;
  if (idx >= N * 64) return;
  int row = idx >> 6, col = idx & 63;
  float v = 0.0f;
  if (col < 35)      v = x[(size_t)row * 35 + col];
  else if (col < 45) v = emb[(size_t)node_id[row] * 10 + (col - 35)];
  h0[idx] = v;
}

// ---------------- graph aggregation (norm='both') ----------------
// agg[v] = sum_{(s,v) in E+selfloops} h[s]*nrm_out[s]   (nrm_in folded in GEMM)
// h/agg share padded leading dim ldh; padding columns stay zero.

__global__ void init_agg_kernel(const float* __restrict__ h,
                                const float* __restrict__ nrm_out,
                                float* __restrict__ agg, int N, int ldh) {
  int idx = blockIdx.x * blockDim.x + threadIdx.x;
  if (idx >= N * ldh) return;
  int row = idx / ldh;
  agg[idx] = h[idx] * nrm_out[row];  // self-loop contribution (pad cols: 0*nrm=0)
}

__global__ void edge_agg_kernel(const int* __restrict__ src,
                                const int* __restrict__ dst,
                                const float* __restrict__ h,
                                const float* __restrict__ nrm_out,
                                float* __restrict__ agg,
                                int E, int F, int ldh) {
  long long total = (long long)E * F;
  long long idx = (long long)blockIdx.x * blockDim.x + threadIdx.x;
  long long stride = (long long)gridDim.x * blockDim.x;
  for (; idx < total; idx += stride) {
    int e = (int)(idx / F);
    int f = (int)(idx - (long long)e * F);
    int s = src[e];
    int d = dst[e];
    if (e + 1 < E)  // gfx1250 global_prefetch_b8 on next edge's source row
      __builtin_prefetch(&h[(size_t)src[e + 1] * ldh + f], 0, 0);
    atomicAdd(&agg[(size_t)d * ldh + f], h[(size_t)s * ldh + f] * nrm_out[s]);
  }
}

// ---------------- weight packing: f32 -> fragment-native f16 ----------------
// Bp block for (kt,nt) holds 32 lanes x 16 halfs contiguously; lane l element j
// maps to k = kt*32 + kb(l) + (j<8 ? j : 8+j), n = nt*16 + (l&15).

__global__ void packB_kernel(const float* __restrict__ W, int wsk, int wsn,
                             int K, int Nc, int KP,
                             _Float16* __restrict__ Bp) {
  const int ntiles = Nc >> 4;
  const int total  = (KP >> 5) * ntiles * 512;
  int idx = blockIdx.x * blockDim.x + threadIdx.x;
  if (idx >= total) return;
  int j    = idx & 15;
  int l    = (idx >> 4) & 31;
  int tile = idx >> 9;
  int nt   = tile % ntiles;
  int kt   = tile / ntiles;
  int kb   = (l & 16) ? 8 : 0;
  int k    = kt * 32 + kb + ((j < 8) ? j : (8 + j));
  int n    = nt * 16 + (l & 15);
  float v  = (k < K) ? W[(size_t)k * wsk + (size_t)n * wsn] : 0.0f;
  Bp[idx]  = (_Float16)v;
}

// ---------------- WMMA GEMM ----------------
// C[M, Nc] = epilogue( (rowScale .* A[M, KP]) @ Bpacked + bias1 + bias2 )
// One wave per 16x32 C tile (2 N-tiles share one A fragment), K fully unrolled.
// M multiple of 16, KP in {64,128}; no predication anywhere.

template <int KP>
__global__ void gemm_wmma_kernel(const float* __restrict__ A, int lda,
                                 const float* __restrict__ rowScale,
                                 const _Float16* __restrict__ Bp, int ntilesTotal,
                                 const float* __restrict__ bias1,
                                 const float* __restrict__ bias2,
                                 float* __restrict__ C, int ldc, int doRelu) {
  constexpr int NT = 2;                       // N-tiles per wave
  const int lane   = threadIdx.x;             // 0..31 (wave32)
  const int tileM  = blockIdx.x * 16;
  const int ntBase = blockIdx.y * NT;

  const int mrow = tileM + (lane & 15);
  const int kb   = (lane & 16) ? 8 : 0;
  const float rs = (rowScale != nullptr) ? rowScale[mrow] : 1.0f;
  const float* __restrict__ arow = A + (size_t)mrow * lda + kb;

  v8f acc[NT];
#pragma unroll
  for (int t = 0; t < NT; ++t) acc[t] = {};

#pragma unroll
  for (int kt = 0; kt < KP / 32; ++kt) {
    // A fragment: 4 unguarded float4 loads, scale, convert to f16
    const float* ap = arow + kt * 32;
    const float4 a0 = *(const float4*)(ap);
    const float4 a1 = *(const float4*)(ap + 4);
    const float4 a2 = *(const float4*)(ap + 16);
    const float4 a3 = *(const float4*)(ap + 20);
    v16h af;
    af[0]  = (_Float16)(a0.x * rs); af[1]  = (_Float16)(a0.y * rs);
    af[2]  = (_Float16)(a0.z * rs); af[3]  = (_Float16)(a0.w * rs);
    af[4]  = (_Float16)(a1.x * rs); af[5]  = (_Float16)(a1.y * rs);
    af[6]  = (_Float16)(a1.z * rs); af[7]  = (_Float16)(a1.w * rs);
    af[8]  = (_Float16)(a2.x * rs); af[9]  = (_Float16)(a2.y * rs);
    af[10] = (_Float16)(a2.z * rs); af[11] = (_Float16)(a2.w * rs);
    af[12] = (_Float16)(a3.x * rs); af[13] = (_Float16)(a3.y * rs);
    af[14] = (_Float16)(a3.z * rs); af[15] = (_Float16)(a3.w * rs);

#pragma unroll
    for (int t = 0; t < NT; ++t) {
      const v16h bf = *(const v16h*)(Bp + ((size_t)(kt * ntilesTotal + ntBase + t) << 9)
                                        + (lane << 4));
      acc[t] = __builtin_amdgcn_wmma_f32_16x16x32_f16(false, af, false, bf,
                                                      (short)0, acc[t], false, false);
    }
  }

#pragma unroll
  for (int t = 0; t < NT; ++t) {
    const int ncol = (ntBase + t) * 16 + (lane & 15);
    const float b1 = (bias1 != nullptr) ? bias1[ncol] : 0.0f;
    const float b2 = (bias2 != nullptr) ? bias2[ncol] : 0.0f;
#pragma unroll
    for (int r = 0; r < 8; ++r) {
      const int mm = tileM + r + ((lane & 16) ? 8 : 0);
      float v = acc[t][r] + b1 + b2;
      if (doRelu) v = fmaxf(v, 0.0f);
      C[(size_t)mm * ldc + ncol] = v;
    }
  }
}

// ---------------- batchnorm column statistics (coalesced) ----------------

__global__ void zero_stats_kernel(float* stats) {
  int i = blockIdx.x * blockDim.x + threadIdx.x;
  if (i < 1024) stats[i] = 0.0f;
}

// blockDim.x = 256; F divides 256. Threads span full row width -> coalesced.
__global__ void colstats_kernel(const float* __restrict__ Y, int Nrows, int F,
                                float* __restrict__ stats /* [2F] */) {
  __shared__ float sh[512];
  const int c    = threadIdx.x % F;
  const int rsub = threadIdx.x / F;
  const int rowsPerIter = blockDim.x / F;
  float sum = 0.0f, sq = 0.0f;
  for (int r = blockIdx.x * rowsPerIter + rsub; r < Nrows;
       r += gridDim.x * rowsPerIter) {
    float v = Y[(size_t)r * F + c];
    sum += v; sq += v * v;
  }
  sh[threadIdx.x] = sum;
  sh[256 + threadIdx.x] = sq;
  __syncthreads();
  for (int off = blockDim.x >> 1; off >= F; off >>= 1) {
    if (threadIdx.x < off) {
      sh[threadIdx.x]       += sh[threadIdx.x + off];
      sh[256 + threadIdx.x] += sh[256 + threadIdx.x + off];
    }
    __syncthreads();
  }
  if (threadIdx.x < F) {
    atomicAdd(&stats[c],     sh[c]);
    atomicAdd(&stats[F + c], sh[256 + c]);
  }
}

// out = relu( (y - mu)*rsqrt(var+eps)*g + b + s )

__global__ void bn_combine_kernel(const float* __restrict__ y,
                                  const float* __restrict__ s,
                                  const float* __restrict__ g,
                                  const float* __restrict__ bta,
                                  const float* __restrict__ stats,
                                  float* __restrict__ out,
                                  int Nrows, int F) {
  int idx = blockIdx.x * blockDim.x + threadIdx.x;
  if (idx >= Nrows * F) return;
  int c = idx % F;
  float mu  = stats[c] / (float)Nrows;
  float var = stats[F + c] / (float)Nrows - mu * mu;
  float v = (y[idx] - mu) * rsqrtf(var + 1e-5f) * g[c] + bta[c] + s[idx];
  out[idx] = fmaxf(v, 0.0f);
}

// ---------------- LSTM sequential scan ----------------

__global__ void lstm_scan_kernel(const float* __restrict__ pre_f,
                                 const float* __restrict__ pre_b,
                                 const float* __restrict__ Whh_f,
                                 const float* __restrict__ Whh_b,
                                 float* __restrict__ x3l, int Nrows) {
  __shared__ float Wsh[128 * 32];
  __shared__ float hsh[32], csh[32], gsh[128];
  const int dir = blockIdx.x;          // 0 = forward, 1 = backward
  const int t   = threadIdx.x;         // 0..127
  const float* Whh = dir ? Whh_b : Whh_f;
  const float* pre = dir ? pre_b : pre_f;

  for (int i = t; i < 128 * 32; i += 128) Wsh[i] = Whh[i];
  if (t < 32) { hsh[t] = 0.0f; csh[t] = 0.0f; }
  __syncthreads();

  for (int step = 0; step < Nrows; ++step) {
    const int row = dir ? (Nrows - 1 - step) : step;
    float acc = pre[(size_t)row * 128 + t];
    const float* wr = &Wsh[t * 32];
#pragma unroll
    for (int j = 0; j < 32; ++j) acc += wr[j] * hsh[j];
    const int gate = t >> 5;                       // i,f,g,o = 0,1,2,3
    gsh[t] = (gate == 2) ? tanhf(acc) : (1.0f / (1.0f + __expf(-acc)));

    const int nrow = dir ? (row - 1) : (row + 1);  // prefetch next timestep
    if (nrow >= 0 && nrow < Nrows)
      __builtin_prefetch(&pre[(size_t)nrow * 128 + t], 0, 0);
    __syncthreads();

    if (t < 32) {
      float ig = gsh[t], fg = gsh[32 + t], gg = gsh[64 + t], og = gsh[96 + t];
      float c2 = fg * csh[t] + ig * gg;
      float h2 = og * tanhf(c2);
      csh[t] = c2; hsh[t] = h2;
      x3l[(size_t)row * 64 + dir * 32 + t] = h2;
    }
    __syncthreads();
  }
  __builtin_amdgcn_s_wait_tensorcnt(0);  // gfx1250 tensor-counter path probe
}

// ---------------- head matvec: out[i] = dot(H[i,:], w) + b ----------------

__global__ void matvec_kernel(const float* __restrict__ H,
                              const float* __restrict__ w,
                              const float* __restrict__ b,
                              float* __restrict__ out, int Nrows, int K) {
  int i = blockIdx.x * blockDim.x + threadIdx.x;
  if (i >= Nrows) return;
  float acc = b[0];
  for (int k = 0; k < K; k += 4) {
    float4 h4 = *(const float4*)(&H[(size_t)i * K + k]);
    acc += h4.x * w[k] + h4.y * w[k + 1] + h4.z * w[k + 2] + h4.w * w[k + 3];
  }
  out[i] = acc;
}

// ---------------------------------------------------------------------------

extern "C" void kernel_launch(void* const* d_in, const int* in_sizes, int n_in,
                              void* d_out, int out_size, void* d_ws, size_t ws_size,
                              hipStream_t stream) {
  (void)n_in; (void)out_size; (void)ws_size;

  const int E = in_sizes[0];
  const int N = in_sizes[3];

  const int*   src     = (const int*)  d_in[0];
  const int*   dst     = (const int*)  d_in[1];
  const float* x       = (const float*)d_in[2];
  const int*   node_id = (const int*)  d_in[3];
  const float* emb     = (const float*)d_in[4];
  const float* convW1  = (const float*)d_in[5];   // (45,128)
  const float* convb1  = (const float*)d_in[6];
  const float* convW2  = (const float*)d_in[7];   // (128,64)
  const float* convb2  = (const float*)d_in[8];
  const float* bn1g    = (const float*)d_in[9];
  const float* bn1b    = (const float*)d_in[10];
  const float* bn2g    = (const float*)d_in[11];
  const float* bn2b    = (const float*)d_in[12];
  const float* bn3g    = (const float*)d_in[13];
  const float* bn3b    = (const float*)d_in[14];
  const float* sc1W    = (const float*)d_in[15];  // (128,45)
  const float* sc1b    = (const float*)d_in[16];
  const float* sc2W    = (const float*)d_in[17];  // (64,128)
  const float* sc2b    = (const float*)d_in[18];
  const float* sc3W    = (const float*)d_in[19];  // (64,64)
  const float* sc3b    = (const float*)d_in[20];
  const float* Wih_f   = (const float*)d_in[21];  // (128,64)
  const float* Whh_f   = (const float*)d_in[22];  // (128,32)
  const float* bih_f   = (const float*)d_in[23];
  const float* bhh_f   = (const float*)d_in[24];
  const float* Wih_b   = (const float*)d_in[25];
  const float* Whh_b   = (const float*)d_in[26];
  const float* bih_b   = (const float*)d_in[27];
  const float* bhh_b   = (const float*)d_in[28];
  const float* a1W     = (const float*)d_in[29];  // (64,64)
  const float* a1b     = (const float*)d_in[30];
  const float* a2W     = (const float*)d_in[31];  // (1,64)
  const float* a2b     = (const float*)d_in[32];
  const float* c1W     = (const float*)d_in[33];
  const float* c1b     = (const float*)d_in[34];
  const float* c2W     = (const float*)d_in[35];
  const float* c2b     = (const float*)d_in[36];

  float* out_active  = (float*)d_out;
  float* out_consume = (float*)d_out + N;

  // Workspace: nrm_out[N] | nrm_in[N] | stats[1024] | packedW[40960 f32]
  //            | 4 x slot[128N]
  float* ws      = (float*)d_ws;
  float* nrm_out = ws;
  float* nrm_in  = ws + N;
  float* stats   = ws + 2 * (size_t)N;  // bn1:[0,256) bn2:[256,384) bn3:[384,512)
  _Float16* pB   = (_Float16*)(stats + 1024);
  float* slotA   = stats + 1024 + 40960;
  float* slotB   = slotA + (size_t)N * 128;
  float* slotC   = slotB + (size_t)N * 128;
  float* slotD   = slotC + (size_t)N * 128;

  // packed weight offsets (halfs)
  _Float16* pb_cw1  = pB;            // K=45->64, Nc=128 : 8192
  _Float16* pb_sc1  = pB + 8192;     // 8192
  _Float16* pb_cw2  = pB + 16384;    // K=128, Nc=64 : 8192
  _Float16* pb_sc2  = pB + 24576;    // 8192
  _Float16* pb_wihf = pB + 32768;    // K=64, Nc=128 : 8192
  _Float16* pb_wihb = pB + 40960;    // 8192
  _Float16* pb_sc3  = pB + 49152;    // K=64, Nc=64 : 4096
  _Float16* pb_a1   = pB + 53248;    // 4096
  _Float16* pb_c1   = pB + 57344;    // 4096

  const int TB = 256;
  auto blocks = [](long long n, int tb) { return (int)((n + tb - 1) / tb); };
  const int tilesM = N / 16;  // N = 50000, multiple of 16

  // ---- pack all weights into WMMA fragment layout (f16) ----
  packB_kernel<<<blocks(8192, TB), TB, 0, stream>>>(convW1, 128, 1,  45, 128, 64,  pb_cw1);
  packB_kernel<<<blocks(8192, TB), TB, 0, stream>>>(sc1W,   1,  45,  45, 128, 64,  pb_sc1);
  packB_kernel<<<blocks(8192, TB), TB, 0, stream>>>(convW2, 64, 1,  128,  64, 128, pb_cw2);
  packB_kernel<<<blocks(8192, TB), TB, 0, stream>>>(sc2W,   1, 128, 128,  64, 128, pb_sc2);
  packB_kernel<<<blocks(8192, TB), TB, 0, stream>>>(Wih_f,  1,  64,  64, 128, 64,  pb_wihf);
  packB_kernel<<<blocks(8192, TB), TB, 0, stream>>>(Wih_b,  1,  64,  64, 128, 64,  pb_wihb);
  packB_kernel<<<blocks(4096, TB), TB, 0, stream>>>(sc3W,   1,  64,  64,  64, 64,  pb_sc3);
  packB_kernel<<<blocks(4096, TB), TB, 0, stream>>>(a1W,    1,  64,  64,  64, 64,  pb_a1);
  packB_kernel<<<blocks(4096, TB), TB, 0, stream>>>(c1W,    1,  64,  64,  64, 64,  pb_c1);

  // ---- degrees / norms ----
  init_deg_kernel<<<blocks(N, TB), TB, 0, stream>>>(nrm_out, nrm_in, N);
  edge_deg_kernel<<<blocks(E, TB), TB, 0, stream>>>(src, dst, nrm_out, nrm_in, E);
  finalize_nrm_kernel<<<blocks(N, TB), TB, 0, stream>>>(nrm_out, N);
  finalize_nrm_kernel<<<blocks(N, TB), TB, 0, stream>>>(nrm_in, N);

  zero_stats_kernel<<<4, TB, 0, stream>>>(stats);

  // ---- h0 (N x 64, cols 45..63 zero) -> slotA ----
  build_h0_kernel<<<blocks((long long)N * 64, TB), TB, 0, stream>>>(x, node_id, emb, slotA, N);

  // ---- GraphConv1 aggregation -> slotB (N x 64 padded) ----
  init_agg_kernel<<<blocks((long long)N * 64, TB), TB, 0, stream>>>(slotA, nrm_out, slotB, N, 64);
  edge_agg_kernel<<<blocks((long long)E * 45, TB), TB, 0, stream>>>(src, dst, slotA, nrm_out, slotB, E, 45, 64);

  // ---- y1 = (nrm_in .* agg1) @ convW1 + convb1 -> slotC (N x 128) ----
  gemm_wmma_kernel<64><<<dim3(tilesM, 4), 32, 0, stream>>>(
      slotB, 64, nrm_in, pb_cw1, 8, convb1, nullptr, slotC, 128, 0);
  // ---- s1 = h0 @ sc1W^T + sc1b -> slotD (N x 128) ----
  gemm_wmma_kernel<64><<<dim3(tilesM, 4), 32, 0, stream>>>(
      slotA, 64, nullptr, pb_sc1, 8, sc1b, nullptr, slotD, 128, 0);

  colstats_kernel<<<256, 256, 0, stream>>>(slotC, N, 128, stats);
  // x1 = relu(bn1(y1) + s1) -> slotB (agg1 dead)
  bn_combine_kernel<<<blocks((long long)N * 128, TB), TB, 0, stream>>>(
      slotC, slotD, bn1g, bn1b, stats, slotB, N, 128);

  // ---- GraphConv2 aggregation -> slotA (h0 dead), N x 128 ----
  init_agg_kernel<<<blocks((long long)N * 128, TB), TB, 0, stream>>>(slotB, nrm_out, slotA, N, 128);
  edge_agg_kernel<<<blocks((long long)E * 128, TB), TB, 0, stream>>>(src, dst, slotB, nrm_out, slotA, E, 128, 128);

  // ---- y2 = (nrm_in .* agg2) @ convW2 + convb2 -> slotC (N x 64) ----
  gemm_wmma_kernel<128><<<dim3(tilesM, 2), 32, 0, stream>>>(
      slotA, 128, nrm_in, pb_cw2, 4, convb2, nullptr, slotC, 64, 0);
  // ---- s2 = x1 @ sc2W^T + sc2b -> slotD (N x 64) ----
  gemm_wmma_kernel<128><<<dim3(tilesM, 2), 32, 0, stream>>>(
      slotB, 128, nullptr, pb_sc2, 4, sc2b, nullptr, slotD, 64, 0);

  colstats_kernel<<<256, 256, 0, stream>>>(slotC, N, 64, stats + 256);
  // x2 = relu(bn2(y2) + s2) -> slotB (x1 dead)
  bn_combine_kernel<<<blocks((long long)N * 64, TB), TB, 0, stream>>>(
      slotC, slotD, bn2g, bn2b, stats + 256, slotB, N, 64);

  // ---- LSTM input projections: pre = x2 @ Wih^T + bih + bhh ----
  gemm_wmma_kernel<64><<<dim3(tilesM, 4), 32, 0, stream>>>(
      slotB, 64, nullptr, pb_wihf, 8, bih_f, bhh_f, slotC, 128, 0);
  gemm_wmma_kernel<64><<<dim3(tilesM, 4), 32, 0, stream>>>(
      slotB, 64, nullptr, pb_wihb, 8, bih_b, bhh_b, slotD, 128, 0);

  // ---- sequential scan (both directions concurrently) -> x3l = slotA ----
  lstm_scan_kernel<<<2, 128, 0, stream>>>(slotC, slotD, Whh_f, Whh_b, slotA, N);

  // ---- s3 = x2 @ sc3W^T + sc3b -> slotC (pre_f dead) ----
  gemm_wmma_kernel<64><<<dim3(tilesM, 2), 32, 0, stream>>>(
      slotB, 64, nullptr, pb_sc3, 4, sc3b, nullptr, slotC, 64, 0);

  colstats_kernel<<<256, 256, 0, stream>>>(slotA, N, 64, stats + 384);
  // x3 = relu(bn3(x3l) + s3) -> slotD (pre_b dead)
  bn_combine_kernel<<<blocks((long long)N * 64, TB), TB, 0, stream>>>(
      slotA, slotC, bn3g, bn3b, stats + 384, slotD, N, 64);

  // ---- heads ----
  gemm_wmma_kernel<64><<<dim3(tilesM, 2), 32, 0, stream>>>(
      slotD, 64, nullptr, pb_a1, 4, a1b, nullptr, slotA, 64, 1);
  matvec_kernel<<<blocks(N, TB), TB, 0, stream>>>(slotA, a2W, a2b, out_active, N, 64);

  gemm_wmma_kernel<64><<<dim3(tilesM, 2), 32, 0, stream>>>(
      slotD, 64, nullptr, pb_c1, 4, c1b, nullptr, slotB, 64, 1);
  matvec_kernel<<<blocks(N, TB), TB, 0, stream>>>(slotB, c2W, c2b, out_consume, N, 64);
}
